// Attention_61005715472422
// MI455X (gfx1250) — matrix operand order repaired and checked by reading, not compiled, not run on hardware
//
#include <hip/hip_runtime.h>
#include <hip/hip_bf16.h>

typedef __bf16 bf16_t;
typedef __attribute__((ext_vector_type(16))) __bf16 v16bf;
typedef __attribute__((ext_vector_type(8)))  __bf16 v8bf;
typedef __attribute__((ext_vector_type(8)))  float  v8f;
typedef __attribute__((ext_vector_type(4)))  unsigned int v4u;
typedef __attribute__((ext_vector_type(4)))  int v4i;
typedef __attribute__((ext_vector_type(8)))  int v8i;

union V16BF { v16bf v; struct { v8bf lo, hi; } p; };

#define WMMA_BF16(A, B, C) \
  __builtin_amdgcn_wmma_f32_16x16x32_bf16(false, (A), false, (B), (short)0, (C), false, false)

#define DIMM 768
#define NTOK 1024
#define HEADS 12
#define HD 64
#define SCALE 0.125f

#ifndef __has_builtin
#define __has_builtin(x) 0
#endif
#if __has_builtin(__builtin_amdgcn_tensor_load_to_lds)
#define HAVE_TDM 1
#else
#define HAVE_TDM 0
#endif

#if HAVE_TDM
// 1D contiguous TDM copy: global -> LDS, nbytes must be a multiple of 8.
// D# bitfields per CDNA5 ISA 8.3/8.4: g0={count=1, lds_addr, global_addr, type=2},
// g1={data_size=8B, tensor_dim0=n, tensor_dim1=1, tile_dim0=n, tile_dim1=1, stride=n}.
__device__ inline void tdm_copy_1d(const void* gsrc, unsigned lds_off, unsigned nbytes) {
  const unsigned n = nbytes >> 3;                      // 8-byte elements
  const unsigned long long ga = (unsigned long long)(size_t)gsrc;
  v4u g0;
  g0[0] = 1u;                                          // count=1, user mode
  g0[1] = lds_off;                                     // lds_addr (bytes)
  g0[2] = (unsigned)(ga & 0xffffffffu);                // global_addr[31:0]
  g0[3] = (unsigned)((ga >> 32) & 0x01ffffffu) | (2u << 30);  // addr[56:32] | type=2
  v8i g1;
  g1[0] = (int)(3u << 16);                             // data_size=3 (8B), no mask/flags
  g1[1] = (int)((n & 0xffffu) << 16);                  // tensor_dim0[15:0]
  g1[2] = (int)((n >> 16) & 0xffffu) | (1 << 16);      // tensor_dim0[31:16] | tensor_dim1=1
  g1[3] = (int)((n & 0xffffu) << 16);                  // tile_dim0 = n
  g1[4] = 1;                                           // tile_dim1=1, tile_dim2=0
  g1[5] = (int)n;                                      // tensor_dim0_stride[31:0]
  g1[6] = 0;                                           // stride hi / dim1_stride lo
  g1[7] = 0;
  v4i z4 = {0, 0, 0, 0};
#if defined(__clang_major__) && __clang_major__ >= 23
  v8i z8 = {0, 0, 0, 0, 0, 0, 0, 0};
  __builtin_amdgcn_tensor_load_to_lds(g0, g1, z4, z4, z8, 0);
#else
  __builtin_amdgcn_tensor_load_to_lds(g0, g1, z4, z4, 0);
#endif
}
#endif

// ---------------- conversion helpers ----------------
__global__ __launch_bounds__(256) void k_cvt(const float* __restrict__ in, bf16_t* __restrict__ out, int n) {
  int i = blockIdx.x * 256 + threadIdx.x;
  if (i < n) out[i] = (bf16_t)in[i];
}

__global__ __launch_bounds__(256) void k_transpose_cvt(const float* __restrict__ in, bf16_t* __restrict__ out,
                                                       int rows, int cols) {
  int i = blockIdx.x * 256 + threadIdx.x;
  if (i < rows * cols) {
    int r = i / cols, c = i - r * cols;
    out[c * rows + r] = (bf16_t)in[i];
  }
}

// ---------------- QKV GEMM: 64x64 per wave (4x4 WMMA frags), 128x256 per block ----------------
__global__ __launch_bounds__(256) void k_qkv_gemm(const bf16_t* __restrict__ X, const bf16_t* __restrict__ Wt,
                                                  const float* __restrict__ bias,
                                                  bf16_t* __restrict__ qb, bf16_t* __restrict__ kb,
                                                  bf16_t* __restrict__ vT) {
  const int lane = threadIdx.x & 31, wid = threadIdx.x >> 5;
  const int m = lane & 15, hf = lane >> 4;
  const int wm = wid >> 2, wn = wid & 3;               // 2x4 wave grid
  const int row0 = blockIdx.x * 128 + wm * 64;
  const int col0 = blockIdx.y * 256 + wn * 64;
  const bf16_t* A0 = X + (size_t)(row0 + m) * DIMM;
  const bf16_t* B0 = Wt + (size_t)(col0 + m) * DIMM;
  v8f acc[4][4] = {};
  for (int k0 = 0; k0 < DIMM; k0 += 32) {
    V16BF a[4];
    v16bf b[4];
#pragma unroll
    for (int i = 0; i < 4; ++i) {
      const bf16_t* ap = A0 + (size_t)i * 16 * DIMM + k0 + hf * 16;
      a[i].p.lo = *(const v8bf*)(ap);
      a[i].p.hi = *(const v8bf*)(ap + 8);
    }
#pragma unroll
    for (int j = 0; j < 4; ++j)
      b[j] = *(const v16bf*)(B0 + (size_t)j * 16 * DIMM + k0 + hf * 16);
#pragma unroll
    for (int i = 0; i < 4; ++i)
#pragma unroll
      for (int j = 0; j < 4; ++j)
        acc[i][j] = WMMA_BF16(a[i].v, b[j], acc[i][j]);
  }
#pragma unroll
  for (int j = 0; j < 4; ++j) {
    const int gcol = col0 + j * 16 + m;
    const float bv = bias[gcol];
    const int which = gcol / DIMM;                     // uniform per 16-tile
    const int hc = gcol - which * DIMM;
    const int head = hc >> 6, c = hc & 63;
#pragma unroll
    for (int i = 0; i < 4; ++i) {
#pragma unroll
      for (int r = 0; r < 8; ++r) {
        const int grow = row0 + i * 16 + r + 8 * hf;
        const int bb = grow >> 10, tok = grow & 1023;
        const int bh = bb * HEADS + head;
        const float val = acc[i][j][r] + bv;
        if (which == 0)      qb[((size_t)bh * NTOK + tok) * HD + c] = (bf16_t)val;
        else if (which == 1) kb[((size_t)bh * NTOK + tok) * HD + c] = (bf16_t)val;
        else                 vT[((size_t)bh * HD + c) * NTOK + tok] = (bf16_t)val;
      }
    }
  }
}

// ---------------- rel_h ----------------
__global__ __launch_bounds__(32) void k_relh(const bf16_t* __restrict__ qb, const bf16_t* __restrict__ rphb,
                                             float* __restrict__ relh) {
  const int bh = blockIdx.x, h = blockIdx.y;
  const int lane = threadIdx.x & 31, m = lane & 15, hf = lane >> 4;
  const bf16_t* Q = qb + ((size_t)bh * NTOK + h * 32) * HD;
  for (int wt = 0; wt < 2; ++wt) {
    const bf16_t* arow = Q + (wt * 16 + m) * HD;
    for (int kt = 0; kt < 2; ++kt) {
      const int kcol = kt * 16 + m;
      const bf16_t* bcol = rphb + (h - kcol + 31) * HD;
      v8f acc = {};
      for (int k0 = 0; k0 < HD; k0 += 32) {
        V16BF a;
        a.p.lo = *(const v8bf*)(arow + k0 + hf * 16);
        a.p.hi = *(const v8bf*)(arow + k0 + 8 + hf * 16);
        v16bf b = *(const v16bf*)(bcol + k0 + hf * 16);
        acc = WMMA_BF16(a.v, b, acc);
      }
#pragma unroll
      for (int r = 0; r < 8; ++r) {
        const int w = wt * 16 + r + 8 * hf;
        relh[(((size_t)bh * 32 + h) * 32 + w) * 32 + kcol] = acc[r];
      }
    }
  }
}

// ---------------- rel_w ----------------
__global__ __launch_bounds__(32) void k_relw(const bf16_t* __restrict__ qb, const bf16_t* __restrict__ rpwb,
                                             float* __restrict__ relw) {
  const int bh = blockIdx.x, w = blockIdx.y;
  const int lane = threadIdx.x & 31, m = lane & 15, hf = lane >> 4;
  const bf16_t* Q = qb + ((size_t)bh * NTOK + w) * HD;
  for (int ht = 0; ht < 2; ++ht) {
    const bf16_t* arow = Q + (size_t)(ht * 16 + m) * 32 * HD;
    for (int kt = 0; kt < 2; ++kt) {
      const int kcol = kt * 16 + m;
      const bf16_t* bcol = rpwb + (w - kcol + 31) * HD;
      v8f acc = {};
      for (int k0 = 0; k0 < HD; k0 += 32) {
        V16BF a;
        a.p.lo = *(const v8bf*)(arow + k0 + hf * 16);
        a.p.hi = *(const v8bf*)(arow + k0 + 8 + hf * 16);
        v16bf b = *(const v16bf*)(bcol + k0 + hf * 16);
        acc = WMMA_BF16(a.v, b, acc);
      }
#pragma unroll
      for (int r = 0; r < 8; ++r) {
        const int hrow = ht * 16 + r + 8 * hf;
        relw[(((size_t)bh * 32 + hrow) * 32 + w) * 32 + kcol] = acc[r];
      }
    }
  }
}

// ---------------- fused attention ----------------
// LDS: S[16][1024] f32 | P[16][1024] bf16 | Qs[16][64] bf16 | RHs[16][32] f32 | RWs[16][32] f32
#define SM_S  0
#define SM_P  (16 * NTOK * 4)
#define SM_Q  (SM_P + 16 * NTOK * 2)
#define SM_RH (SM_Q + 16 * HD * 2)
#define SM_RW (SM_RH + 16 * 32 * 4)
#define SM_SZ (SM_RW + 16 * 32 * 4)

__global__ __launch_bounds__(256) void k_attn(const bf16_t* __restrict__ qb, const bf16_t* __restrict__ kb,
                                              const bf16_t* __restrict__ vT, const float* __restrict__ relh,
                                              const float* __restrict__ relw, bf16_t* __restrict__ aout) {
  extern __shared__ char smem[];
  float*  S   = (float*)(smem + SM_S);
  bf16_t* P   = (bf16_t*)(smem + SM_P);
  bf16_t* Qs  = (bf16_t*)(smem + SM_Q);
  float*  RHs = (float*)(smem + SM_RH);
  float*  RWs = (float*)(smem + SM_RW);
  const int lane = threadIdx.x & 31, wid = threadIdx.x >> 5;
  const int m = lane & 15, hf = lane >> 4;
  const int q0 = blockIdx.x * 16, bh = blockIdx.y;

  const bf16_t* qsrc  = qb + ((size_t)bh * NTOK + q0) * HD;           // 16x64 contiguous
  const float*  rhsrc = relh + (size_t)bh * 32 * 32 * 32 + q0 * 32;   // 16x32 contiguous
  const float*  rwsrc = relw + (size_t)bh * 32 * 32 * 32 + q0 * 32;

#if HAVE_TDM
  if (wid == 0) {       // TDM DMA: stage Q tile + rel slabs, one wave issues, EXEC-independent
    const unsigned base = (unsigned)(size_t)(void*)smem;
    tdm_copy_1d(qsrc,  base + SM_Q,  16 * HD * 2);
    tdm_copy_1d(rhsrc, base + SM_RH, 16 * 32 * 4);
    tdm_copy_1d(rwsrc, base + SM_RW, 16 * 32 * 4);
    __builtin_amdgcn_s_wait_tensorcnt(0);
  }
#else
  for (int i = threadIdx.x; i < 16 * HD; i += 256) Qs[i] = qsrc[i];
  for (int i = threadIdx.x; i < 16 * 32; i += 256) { RHs[i] = rhsrc[i]; RWs[i] = rwsrc[i]; }
#endif
  __syncthreads();

  // A fragments of the Q tile from LDS (reused across all 64 key tiles)
  const bf16_t* Qr = Qs + m * HD;
  V16BF a0, a1;
  a0.p.lo = *(const v8bf*)(Qr + hf * 16);       a0.p.hi = *(const v8bf*)(Qr + 8 + hf * 16);
  a1.p.lo = *(const v8bf*)(Qr + 32 + hf * 16);  a1.p.hi = *(const v8bf*)(Qr + 40 + hf * 16);

  for (int kt = wid; kt < 64; kt += 8) {
    const int key = kt * 16 + m;
    const bf16_t* bcol = kb + ((size_t)bh * NTOK + key) * HD;
    v8f acc = {};
    v16bf b0 = *(const v16bf*)(bcol + hf * 16);
    acc = WMMA_BF16(a0.v, b0, acc);
    v16bf b1 = *(const v16bf*)(bcol + 32 + hf * 16);
    acc = WMMA_BF16(a1.v, b1, acc);
    const int kh = key >> 5, kw = key & 31;
#pragma unroll
    for (int r = 0; r < 8; ++r) {
      const int M = r + 8 * hf;
      S[M * NTOK + key] = acc[r] * SCALE + RHs[M * 32 + kh] + RWs[M * 32 + kw];
    }
  }
  __syncthreads();

  for (int row = wid; row < 16; row += 8) {
    float* Srow = S + row * NTOK;
    float mx = -3.4e38f;
    for (int j = lane; j < NTOK; j += 32) mx = fmaxf(mx, Srow[j]);
#pragma unroll
    for (int off = 16; off > 0; off >>= 1) mx = fmaxf(mx, __shfl_xor(mx, off, 32));
    float sum = 0.f;
    for (int j = lane; j < NTOK; j += 32) { float e = __expf(Srow[j] - mx); Srow[j] = e; sum += e; }
#pragma unroll
    for (int off = 16; off > 0; off >>= 1) sum += __shfl_xor(sum, off, 32);
    const float inv = 1.0f / sum;
    bf16_t* Prow = P + row * NTOK;
    for (int j = lane; j < NTOK; j += 32) Prow[j] = (bf16_t)(Srow[j] * inv);
  }
  __syncthreads();

  if (wid < 4) {
    const int cch = wid * 16 + m;
    const bf16_t* vcol = vT + ((size_t)bh * HD + cch) * NTOK;
    const bf16_t* pa = P + m * NTOK;
    v8f acc = {};
    for (int k0 = 0; k0 < NTOK; k0 += 32) {
      V16BF a;
      a.p.lo = *(const v8bf*)(pa + k0 + hf * 16);
      a.p.hi = *(const v8bf*)(pa + k0 + 8 + hf * 16);
      v16bf b = *(const v16bf*)(vcol + k0 + hf * 16);
      acc = WMMA_BF16(a.v, b, acc);
    }
    const int bb = bh / HEADS, head = bh - bb * HEADS;
#pragma unroll
    for (int r = 0; r < 8; ++r) {
      const int tok = q0 + r + 8 * hf;
      aout[((size_t)bb * NTOK + tok) * DIMM + head * HD + cch] = (bf16_t)acc[r];
    }
  }
}

// ---------------- proj GEMM: 64x64 per wave (4x4 frags), fp32 out + bias ----------------
__global__ __launch_bounds__(256) void k_proj(const bf16_t* __restrict__ A, const bf16_t* __restrict__ Wt,
                                              const float* __restrict__ bias, float* __restrict__ out) {
  const int lane = threadIdx.x & 31, wid = threadIdx.x >> 5;
  const int m = lane & 15, hf = lane >> 4;
  const int wm = wid >> 2, wn = wid & 3;
  const int row0 = blockIdx.x * 128 + wm * 64;
  const int col0 = blockIdx.y * 256 + wn * 64;
  const bf16_t* A0 = A + (size_t)(row0 + m) * DIMM;
  const bf16_t* B0 = Wt + (size_t)(col0 + m) * DIMM;
  v8f acc[4][4] = {};
  for (int k0 = 0; k0 < DIMM; k0 += 32) {
    V16BF a[4];
    v16bf b[4];
#pragma unroll
    for (int i = 0; i < 4; ++i) {
      const bf16_t* ap = A0 + (size_t)i * 16 * DIMM + k0 + hf * 16;
      a[i].p.lo = *(const v8bf*)(ap);
      a[i].p.hi = *(const v8bf*)(ap + 8);
    }
#pragma unroll
    for (int j = 0; j < 4; ++j)
      b[j] = *(const v16bf*)(B0 + (size_t)j * 16 * DIMM + k0 + hf * 16);
#pragma unroll
    for (int i = 0; i < 4; ++i)
#pragma unroll
      for (int j = 0; j < 4; ++j)
        acc[i][j] = WMMA_BF16(a[i].v, b[j], acc[i][j]);
  }
#pragma unroll
  for (int j = 0; j < 4; ++j) {
    const float bv = bias[col0 + j * 16 + m];
#pragma unroll
    for (int i = 0; i < 4; ++i)
#pragma unroll
      for (int r = 0; r < 8; ++r)
        out[(size_t)(row0 + i * 16 + r + 8 * hf) * DIMM + col0 + j * 16 + m] = acc[i][j][r] + bv;
  }
}

extern "C" void kernel_launch(void* const* d_in, const int* in_sizes, int n_in,
                              void* d_out, int out_size, void* d_ws, size_t ws_size,
                              hipStream_t stream) {
  (void)in_sizes; (void)n_in; (void)out_size; (void)ws_size;
  const float* x      = (const float*)d_in[0];
  const float* qkv_w  = (const float*)d_in[1];
  const float* qkv_b  = (const float*)d_in[2];
  const float* proj_w = (const float*)d_in[3];
  const float* proj_b = (const float*)d_in[4];
  const float* rph    = (const float*)d_in[5];
  const float* rpw    = (const float*)d_in[6];
  float* out = (float*)d_out;

  char* ws = (char*)d_ws;
  size_t off = 0;
  auto alloc = [&](size_t bytes) -> char* {
    char* p = ws + off;
    off = (off + bytes + 255) & ~(size_t)255;
    return p;
  };
  bf16_t* xb     = (bf16_t*)alloc((size_t)8192 * DIMM * 2);
  bf16_t* wqkvT  = (bf16_t*)alloc((size_t)2304 * DIMM * 2);
  bf16_t* wprojT = (bf16_t*)alloc((size_t)DIMM * DIMM * 2);
  bf16_t* rphb   = (bf16_t*)alloc((size_t)63 * HD * 2);
  bf16_t* rpwb   = (bf16_t*)alloc((size_t)63 * HD * 2);
  bf16_t* qbuf   = (bf16_t*)alloc((size_t)96 * NTOK * HD * 2);
  bf16_t* kbuf   = (bf16_t*)alloc((size_t)96 * NTOK * HD * 2);
  bf16_t* vTbuf  = (bf16_t*)alloc((size_t)96 * HD * NTOK * 2);
  float*  relh   = (float*)alloc((size_t)96 * 32 * 32 * 32 * 4);
  float*  relw   = (float*)alloc((size_t)96 * 32 * 32 * 32 * 4);
  bf16_t* aout   = (bf16_t*)alloc((size_t)8192 * DIMM * 2);

  k_cvt<<<(8192 * DIMM + 255) / 256, 256, 0, stream>>>(x, xb, 8192 * DIMM);
  k_transpose_cvt<<<(DIMM * 2304 + 255) / 256, 256, 0, stream>>>(qkv_w, wqkvT, DIMM, 2304);
  k_transpose_cvt<<<(DIMM * DIMM + 255) / 256, 256, 0, stream>>>(proj_w, wprojT, DIMM, DIMM);
  k_cvt<<<(63 * HD + 255) / 256, 256, 0, stream>>>(rph, rphb, 63 * HD);
  k_cvt<<<(63 * HD + 255) / 256, 256, 0, stream>>>(rpw, rpwb, 63 * HD);

  k_qkv_gemm<<<dim3(64, 9), 256, 0, stream>>>(xb, wqkvT, qkv_b, qbuf, kbuf, vTbuf);
  k_relh<<<dim3(96, 32), 32, 0, stream>>>(qbuf, rphb, relh);
  k_relw<<<dim3(96, 32), 32, 0, stream>>>(qbuf, rpwb, relw);

  (void)hipFuncSetAttribute((const void*)k_attn, hipFuncAttributeMaxDynamicSharedMemorySize, SM_SZ);
  k_attn<<<dim3(64, 96), 256, SM_SZ, stream>>>(qbuf, kbuf, vTbuf, relh, relw, aout);

  k_proj<<<dim3(64, 3), 256, 0, stream>>>(aout, wprojT, proj_b, out);
}
